// PolicyValueNet_32469952758037
// MI455X (gfx1250) — matrix operand order repaired and checked by reading, not compiled, not run on hardware
//
#include <hip/hip_runtime.h>

// ---------------------------------------------------------------------------
// Types for CDNA5 WMMA (wave32)
// ---------------------------------------------------------------------------
typedef __bf16 bf16_t;
typedef __bf16 v16bf __attribute__((ext_vector_type(16)));
typedef __bf16 v4bf  __attribute__((ext_vector_type(4)));
typedef float  v8f   __attribute__((ext_vector_type(8)));

// ---------------------------------------------------------------------------
// Degree / normalization
// ---------------------------------------------------------------------------
__global__ void k_deg_init(float* deg, int n) {
    int i = blockIdx.x * blockDim.x + threadIdx.x;
    if (i < n) deg[i] = 1.0f;                       // self-loop contribution
}

__global__ void k_deg_accum(const int* __restrict__ dst, float* deg, int e) {
    int i = blockIdx.x * blockDim.x + threadIdx.x;
    if (i < e)
        __hip_atomic_fetch_add(&deg[dst[i]], 1.0f,
                               __ATOMIC_RELAXED, __HIP_MEMORY_SCOPE_AGENT);
}

__global__ void k_rsqrt_inplace(float* d, int n) {
    int i = blockIdx.x * blockDim.x + threadIdx.x;
    if (i < n) d[i] = rsqrtf(d[i]);
}

// ---------------------------------------------------------------------------
// f32 -> bf16 row-major conversion (x -> hbf), 4 elements / thread
// ---------------------------------------------------------------------------
__global__ void k_f32_to_bf16(const float* __restrict__ in, bf16_t* __restrict__ out, int n4) {
    int i = blockIdx.x * blockDim.x + threadIdx.x;
    if (i >= n4) return;
    float4 v = ((const float4*)in)[i];
    v4bf o;
    o[0] = (__bf16)v.x; o[1] = (__bf16)v.y; o[2] = (__bf16)v.z; o[3] = (__bf16)v.w;
    ((v4bf*)out)[i] = o;
}

// ---------------------------------------------------------------------------
// Weight prep: f32 row-major [K=128][N=128] -> bf16 column-major [N][K]
// ---------------------------------------------------------------------------
__global__ void k_wprep(const float* __restrict__ W, bf16_t* __restrict__ Wt) {
    int i = blockIdx.x * blockDim.x + threadIdx.x;
    if (i >= 128 * 128) return;
    int k = i >> 7, n = i & 127;
    Wt[n * 128 + k] = (__bf16)W[k * 128 + n];
}

// ---------------------------------------------------------------------------
// WMMA GEMM: out[rows,128] = A(bf16,[rows,128]) @ W(128x128)
//   A fragments per ISA 16-bit A 16x32 layout; B from LDS (col-major weights).
//   fuse==1: out_bf = bf16(relu(acc + bias));  fuse==0: out_f = acc (f32)
//   One wave per 16-row strip; 8 waves / block; 32 v_wmma per wave.
// ---------------------------------------------------------------------------
__global__ __launch_bounds__(256) void k_gemm_wmma(
    const bf16_t* __restrict__ A, const bf16_t* __restrict__ Wt,
    float* __restrict__ outf, bf16_t* __restrict__ outb,
    const float* __restrict__ bias, int fuse)
{
    __shared__ __align__(16) bf16_t sW[128 * 128];   // 32 KB of 320 KB LDS
    {
        const uint4* g = (const uint4*)Wt;
        uint4* s = (uint4*)sW;
        for (int i = threadIdx.x; i < 2048; i += 256) s[i] = g[i];
    }
    __syncthreads();

    const int wave = threadIdx.x >> 5;
    const int lane = threadIdx.x & 31;
    const int lrow = lane & 15;
    const int lhi  = lane >> 4;
    const int row_base = (blockIdx.x * 8 + wave) * 16;

    // A fragments: lane<16 holds row lrow K={k0..k0+7, k0+16..k0+23}, k0 = kt*32 (+8 for hi half)
    const bf16_t* arow = A + (size_t)(row_base + lrow) * 128;
    v16bf afrag[4];
#pragma unroll
    for (int kt = 0; kt < 4; ++kt) {
        const int k0 = kt * 32 + lhi * 8;
        v16bf a;
        *(uint4*)&a         = *(const uint4*)(arow + k0);
        *(((uint4*)&a) + 1) = *(const uint4*)(arow + k0 + 16);
        afrag[kt] = a;
    }

#pragma unroll
    for (int nt = 0; nt < 8; ++nt) {
        v8f c = {0.f, 0.f, 0.f, 0.f, 0.f, 0.f, 0.f, 0.f};
        // B fragment: lane -> column (nt*16+lrow), K={k0..k0+7, k0+16..k0+23}, k0 = kt*32 + lhi*8
        const bf16_t* wcol = sW + (nt * 16 + lrow) * 128 + lhi * 8;
#pragma unroll
        for (int kt = 0; kt < 4; ++kt) {
            v16bf b;
            *(uint4*)&b         = *(const uint4*)(wcol + kt * 32);
            *(((uint4*)&b) + 1) = *(const uint4*)(wcol + kt * 32 + 16);
            c = __builtin_amdgcn_wmma_f32_16x16x32_bf16(
                    false, afrag[kt], false, b, (short)0, c, false, false);
        }
        // C/D layout: VGPR r -> M = r + 8*lhi, N = lane&15
        const int orow = row_base + lhi * 8;
        const int ocol = nt * 16 + lrow;
        if (fuse) {
            const float bb = bias[ocol];
#pragma unroll
            for (int r = 0; r < 8; ++r) {
                float v = c[r] + bb;
                outb[(size_t)(orow + r) * 128 + ocol] = (__bf16)(v > 0.f ? v : 0.f);
            }
        } else {
#pragma unroll
            for (int r = 0; r < 8; ++r)
                outf[(size_t)(orow + r) * 128 + ocol] = c[r];
        }
    }
}

// ---------------------------------------------------------------------------
// agg[n][:] = t[n][:] * dis[n]^2    (self-loop term; also initializes agg)
// ---------------------------------------------------------------------------
__global__ void k_selfloop_init(const float* __restrict__ t, const float* __restrict__ dis,
                                float* __restrict__ agg, int n) {
    int g = blockIdx.x * blockDim.x + threadIdx.x;   // n*32 threads, 4 floats each
    int node = g >> 5;
    if (node >= n) return;
    int c = (g & 31) * 4;
    float w = dis[node]; w *= w;
    float4 v = *(const float4*)(t + (size_t)node * 128 + c);
    v.x *= w; v.y *= w; v.z *= w; v.w *= w;
    *(float4*)(agg + (size_t)node * 128 + c) = v;
}

// ---------------------------------------------------------------------------
// Edge scatter: one wave per edge; lane handles 4 consecutive floats.
// Gather t[src] (L2-resident, 16B loads), scale by dis[s]*dis[d],
// hardware global_atomic_add_f32 into agg[dst].
// ---------------------------------------------------------------------------
__global__ __launch_bounds__(256) void k_edge_scatter(
    const int* __restrict__ src, const int* __restrict__ dst,
    const float* __restrict__ dis, const float* __restrict__ t,
    float* __restrict__ agg, int e)
{
    int g = blockIdx.x * blockDim.x + threadIdx.x;
    int eid = g >> 5;
    if (eid >= e) return;
    int lane = threadIdx.x & 31;
    int s = src[eid], d = dst[eid];
    float w = dis[s] * dis[d];
    float4 v = *(const float4*)(t + (size_t)s * 128 + lane * 4);
    float* o = agg + (size_t)d * 128 + lane * 4;
    __hip_atomic_fetch_add(o + 0, v.x * w, __ATOMIC_RELAXED, __HIP_MEMORY_SCOPE_AGENT);
    __hip_atomic_fetch_add(o + 1, v.y * w, __ATOMIC_RELAXED, __HIP_MEMORY_SCOPE_AGENT);
    __hip_atomic_fetch_add(o + 2, v.z * w, __ATOMIC_RELAXED, __HIP_MEMORY_SCOPE_AGENT);
    __hip_atomic_fetch_add(o + 3, v.w * w, __ATOMIC_RELAXED, __HIP_MEMORY_SCOPE_AGENT);
}

// ---------------------------------------------------------------------------
// hbf[n][:] = bf16(relu(agg[n][:] + bias))
// ---------------------------------------------------------------------------
__global__ void k_finalize(const float* __restrict__ agg, const float* __restrict__ bias,
                           bf16_t* __restrict__ hbf, int n) {
    int g = blockIdx.x * blockDim.x + threadIdx.x;
    int node = g >> 5;
    if (node >= n) return;
    int c = (g & 31) * 4;
    float4 v = *(const float4*)(agg + (size_t)node * 128 + c);
    float4 b = *(const float4*)(bias + c);
    v4bf o;
    o[0] = (__bf16)fmaxf(v.x + b.x, 0.f);
    o[1] = (__bf16)fmaxf(v.y + b.y, 0.f);
    o[2] = (__bf16)fmaxf(v.z + b.z, 0.f);
    o[3] = (__bf16)fmaxf(v.w + b.w, 0.f);
    *(v4bf*)(hbf + (size_t)node * 128 + c) = o;
}

// ---------------------------------------------------------------------------
// Mean pool: one block per graph, thread d sums column d over npg nodes.
// ---------------------------------------------------------------------------
__global__ void k_pool(const bf16_t* __restrict__ hbf, float* __restrict__ gout, int npg) {
    int b = blockIdx.x, d = threadIdx.x;             // 128 threads
    const bf16_t* p = hbf + (size_t)b * npg * 128 + d;
    float s = 0.f;
    for (int i = 0; i < npg; ++i) s += (float)p[(size_t)i * 128];
    gout[b * 128 + d] = s / (float)npg;
}

// ---------------------------------------------------------------------------
// Value head: values[b] = relu(g[b]@Wv1+bv1) @ Wv2 + bv2
// ---------------------------------------------------------------------------
__global__ void k_value_head(const float* __restrict__ g,
                             const float* __restrict__ Wv1, const float* __restrict__ bv1,
                             const float* __restrict__ Wv2, const float* __restrict__ bv2,
                             float* __restrict__ out_vals) {
    int b = blockIdx.x, j = threadIdx.x;             // 128 threads
    __shared__ float sg[128];
    __shared__ float red[128];
    sg[j] = g[b * 128 + j];
    __syncthreads();
    float acc = bv1[j];
    for (int k = 0; k < 128; ++k) acc += sg[k] * Wv1[k * 128 + j];
    acc = fmaxf(acc, 0.f);
    red[j] = acc * Wv2[j];
    __syncthreads();
    for (int s = 64; s > 0; s >>= 1) {
        if (j < s) red[j] += red[j + s];
        __syncthreads();
    }
    if (j == 0) out_vals[b] = red[0] + bv2[0];
}

// ---------------------------------------------------------------------------
// Node head projection: out[n*8+a] = u[n]@Wn2[:,a] + bn2[a]
// One wave per node; lane covers 4 K values; wave xor-reduction.
// (Packed logits for equal-size sorted graphs == node-major flat layout.)
// ---------------------------------------------------------------------------
__global__ __launch_bounds__(256) void k_node_out(
    const bf16_t* __restrict__ u, const float* __restrict__ Wn2,
    const float* __restrict__ bn2, float* __restrict__ out, int n)
{
    __shared__ float sw[128 * 8];
    __shared__ float sb[8];
    for (int i = threadIdx.x; i < 1024; i += 256) sw[i] = Wn2[i];
    if (threadIdx.x < 8) sb[threadIdx.x] = bn2[threadIdx.x];
    __syncthreads();
    int g = blockIdx.x * blockDim.x + threadIdx.x;
    int node = g >> 5;
    if (node >= n) return;
    int lane = threadIdx.x & 31;
    v4bf v = *(const v4bf*)(u + (size_t)node * 128 + lane * 4);
    float f0 = (float)v[0], f1 = (float)v[1], f2 = (float)v[2], f3 = (float)v[3];
    int k = lane * 4;
    float p[8];
#pragma unroll
    for (int a = 0; a < 8; ++a)
        p[a] = f0 * sw[(k + 0) * 8 + a] + f1 * sw[(k + 1) * 8 + a] +
               f2 * sw[(k + 2) * 8 + a] + f3 * sw[(k + 3) * 8 + a];
#pragma unroll
    for (int off = 16; off > 0; off >>= 1)
#pragma unroll
        for (int a = 0; a < 8; ++a)
            p[a] += __shfl_xor(p[a], off, 32);
    if (lane < 8)
        out[(size_t)node * 8 + lane] = p[lane] + sb[lane];
}

// ---------------------------------------------------------------------------
// Host-side orchestration (all on `stream`; scratch only from d_ws)
// ---------------------------------------------------------------------------
extern "C" void kernel_launch(void* const* d_in, const int* in_sizes, int n_in,
                              void* d_out, int out_size, void* d_ws, size_t ws_size,
                              hipStream_t stream)
{
    const float* x   = (const float*)d_in[0];
    const int*   ei  = (const int*)d_in[1];
    const float* W1  = (const float*)d_in[3];
    const float* b1  = (const float*)d_in[4];
    const float* W2  = (const float*)d_in[5];
    const float* b2  = (const float*)d_in[6];
    const float* W3  = (const float*)d_in[7];
    const float* b3  = (const float*)d_in[8];
    const float* Wn1 = (const float*)d_in[9];
    const float* bn1 = (const float*)d_in[10];
    const float* Wn2 = (const float*)d_in[11];
    const float* bn2 = (const float*)d_in[12];
    const float* Wv1 = (const float*)d_in[13];
    const float* bv1 = (const float*)d_in[14];
    const float* Wv2 = (const float*)d_in[15];
    const float* bv2 = (const float*)d_in[16];

    const int N = in_sizes[0] / 128;
    const int E = in_sizes[1] / 2;
    const int B = 64;
    const int* src = ei;
    const int* dst = ei + E;

    // Workspace carve (256B aligned slices)
    char* ws = (char*)d_ws;
    auto carve = [&](size_t bytes) -> char* {
        char* p = ws;
        ws += (bytes + 255) & ~(size_t)255;
        return p;
    };
    float*  dis   = (float*) carve((size_t)N * 4);
    bf16_t* hbf   = (bf16_t*)carve((size_t)N * 128 * 2);
    float*  t     = (float*) carve((size_t)N * 128 * 4);
    float*  agg   = (float*) carve((size_t)N * 128 * 4);
    bf16_t* Wt0   = (bf16_t*)carve(128 * 128 * 2);
    bf16_t* Wt1   = (bf16_t*)carve(128 * 128 * 2);
    bf16_t* Wt2   = (bf16_t*)carve(128 * 128 * 2);
    bf16_t* Wt3   = (bf16_t*)carve(128 * 128 * 2);
    float*  gpool = (float*) carve((size_t)B * 128 * 4);
    bf16_t* u     = (bf16_t*)t;                   // reuse t after layer 3

    const dim3 blk(256);

    // 1) symmetric normalization: dis = rsqrt(indeg + 1)
    k_deg_init   <<<(N + 255) / 256, blk, 0, stream>>>(dis, N);
    k_deg_accum  <<<(E + 255) / 256, blk, 0, stream>>>(dst, dis, E);
    k_rsqrt_inplace<<<(N + 255) / 256, blk, 0, stream>>>(dis, N);

    // 2) input + weights to bf16 (weights transposed to col-major for WMMA B)
    k_f32_to_bf16<<<(N * 128 / 4 + 255) / 256, blk, 0, stream>>>(x, hbf, N * 128 / 4);
    k_wprep<<<64, blk, 0, stream>>>(W1,  Wt0);
    k_wprep<<<64, blk, 0, stream>>>(W2,  Wt1);
    k_wprep<<<64, blk, 0, stream>>>(W3,  Wt2);
    k_wprep<<<64, blk, 0, stream>>>(Wn1, Wt3);

    // 3) three GCN layers
    const float*  biases[3] = {b1, b2, b3};
    const bf16_t* wts[3]    = {Wt0, Wt1, Wt2};
    const int gemm_blocks = N / (16 * 8);                 // 16 rows/wave, 8 waves/block
    const int row_threads = N * 32;                       // 1 wave per node row
    for (int l = 0; l < 3; ++l) {
        k_gemm_wmma    <<<gemm_blocks, blk, 0, stream>>>(hbf, wts[l], t, nullptr, nullptr, 0);
        k_selfloop_init<<<(row_threads + 255) / 256, blk, 0, stream>>>(t, dis, agg, N);
        k_edge_scatter <<<(E * 32 + 255) / 256, blk, 0, stream>>>(src, dst, dis, t, agg, E);
        k_finalize     <<<(row_threads + 255) / 256, blk, 0, stream>>>(agg, biases[l], hbf, N);
    }

    // 4) node head: u = relu(h @ Wn1 + bn1) (fused in GEMM), then @ Wn2 + bn2
    k_gemm_wmma<<<gemm_blocks, blk, 0, stream>>>(hbf, Wt3, nullptr, u, bn1, 1);
    float* out_logits = (float*)d_out;                    // [B, MAX_A] == [N, 8] flat
    float* out_values = out_logits + (size_t)N * 8;
    k_node_out<<<(N * 32 + 255) / 256, blk, 0, stream>>>(u, Wn2, bn2, out_logits, N);

    // 5) pooled value head
    k_pool      <<<B, dim3(128), 0, stream>>>(hbf, gpool, N / B);
    k_value_head<<<B, dim3(128), 0, stream>>>(gpool, Wv1, bv1, Wv2, bv2, out_values);
}
